// GraphNetRegressor_18056042512461
// MI455X (gfx1250) — matrix operand order repaired and checked by reading, not compiled, not run on hardware
//
#include <hip/hip_runtime.h>
#include <hip/hip_bf16.h>

// ---------------------------------------------------------------------------
// GraphNetRegressor on MI455X (gfx1250), wave32 + WMMA f16 16x16x32.
// B=512, S=512, H=128, L=4. ~310 GFLOP, ~2 GB HBM traffic -> balanced at
// 23.3 TB/s vs f16 WMMA peak. f16 WMMA GEMMs w/ f32 accumulation, f32
// residual stream in HBM, weights staged in LDS via async-to-LDS copies.
// ---------------------------------------------------------------------------

#define Bn   512
#define Sn   512
#define Hn   128
#define Ln   4

typedef __attribute__((ext_vector_type(16))) _Float16 v16h;
typedef __attribute__((ext_vector_type(8)))  _Float16 v8h;
typedef __attribute__((ext_vector_type(8)))  float    v8f;
typedef __attribute__((ext_vector_type(4)))  float    v4f;

static __device__ inline v8f wmma16(v16h a, v16h b, v8f c) {
    return __builtin_amdgcn_wmma_f32_16x16x32_f16(
        false, a, false, b, (short)0, c, false, false);
}
static __device__ inline v16h make_a(v8h lo, v8h hi) {
    return __builtin_shufflevector(lo, hi, 0,1,2,3,4,5,6,7,8,9,10,11,12,13,14,15);
}
// A 16x32 f16 fragment from a row-major f16 row: lanes 0-15 take K {0..7,16..23},
// lanes 16-31 take K {8..15,24..31} (ISA 7.12.2) -> two contiguous 16B loads.
static __device__ inline v16h load_a(const _Float16* src, int koff8) {
    v8h lo = *(const v8h*)(src + koff8);
    v8h hi = *(const v8h*)(src + 16 + koff8);
    return make_a(lo, hi);
}

// LDS layout (halves)
constexpr int LW1_H   = 256 * 128;           // frag-ordered K=256 weight
constexpr int LW2_H   = 128 * 128;           // frag-ordered K=128 weight
constexpr int TSTRIDE = 136;                 // 128 + 8 pad (bank spread)
constexpr int T_H     = 8 * 16 * TSTRIDE;    // per-wave 16x128 bounce tile
constexpr int SMEM_BYTES = (LW1_H + LW2_H + T_H) * 2 + 2 * 128 * 4;

// ---- CDNA5-specific sync / async helpers ----------------------------------
static __device__ inline void lds_fence() {
    asm volatile("s_wait_dscnt 0x0" ::: "memory");
}
static __device__ inline void async_fence() {
    asm volatile("s_wait_asynccnt 0x0" ::: "memory");
}
// LDS aperture: flat address low 32 bits are the LDS byte offset (ISA 10.2).
static __device__ inline unsigned lds_addr(const void* p) {
    return (unsigned)(unsigned long long)p;
}
// GVS-form async copy: 16B global -> LDS, tracked by ASYNCcnt (ISA 08 §4).
static __device__ inline void async_ld16(unsigned lds_off, const void* sbase,
                                         int voff) {
    asm volatile("global_load_async_to_lds_b128 %0, %1, %2 offset:0"
                 :: "v"(lds_off), "v"(voff), "s"(sbase)
                 : "memory");
}

// ---------------------------------------------------------------------------
// Weight conversion: f32 row-major [K][128] -> f16 WMMA-B fragment order.
// Fragment elem j of (ks,ct,lane): k = ks*32 + (lane>=16?8:0) + j + (j>=8?8:0),
//                                  n = ct*16 + (lane&15)
// ---------------------------------------------------------------------------
__global__ void wfrag_kernel(const float* __restrict__ src,
                             _Float16* __restrict__ dst, int K, int nmat) {
    int tid   = blockIdx.x * blockDim.x + threadIdx.x;
    int units = nmat * (K / 32) * 8 * 32;
    if (tid >= units) return;
    int lane = tid & 31;  int r = tid >> 5;
    int ct = r & 7;       r >>= 3;
    int ks = r % (K / 32);
    int mat = r / (K / 32);
    int n  = ct * 16 + (lane & 15);
    int kb = ks * 32 + ((lane & 16) ? 8 : 0);
    const float* s = src + (size_t)mat * K * 128;
    _Float16*    d = dst + (size_t)mat * K * 128 + (((ks * 8 + ct) * 32 + lane) << 4);
#pragma unroll
    for (int j = 0; j < 16; ++j)
        d[j] = (_Float16)s[(size_t)(kb + j + (j >= 8 ? 8 : 0)) * 128 + n];
}

// ---------------------------------------------------------------------------
__global__ void embed_kernel(const int* __restrict__ tok,
                             const float* __restrict__ mask,
                             const float* __restrict__ aa,
                             const float* __restrict__ pos,
                             float* __restrict__ h32,
                             _Float16* __restrict__ h16) {
    int tid = blockIdx.x * blockDim.x + threadIdx.x;
    if (tid >= Bn * Sn * (Hn / 4)) return;
    int k4 = (tid & (Hn / 4 - 1)) * 4;
    int bs = tid / (Hn / 4);
    int s  = bs & (Sn - 1);
    int t  = tok[bs];
    float m = mask[bs];
    size_t base = (size_t)bs * Hn + k4;
#pragma unroll
    for (int j = 0; j < 4; ++j) {
        float v = (aa[(size_t)t * Hn + k4 + j] + pos[(size_t)s * Hn + k4 + j]) * m;
        h32[base + j] = v;
        h16[base + j] = (_Float16)v;
    }
}

// ---------------------------------------------------------------------------
// LDS fill: async global->LDS for W1/W2 fragments, VALU-free; drain ASYNCcnt.
// ---------------------------------------------------------------------------
static __device__ inline void fill_lds(const _Float16* gW1, const _Float16* gW2,
                                       const float* gB1, const float* gB2,
                                       _Float16* lW1, _Float16* lW2,
                                       float* lB1, float* lB2, int tid) {
    unsigned l1 = lds_addr(lW1), l2 = lds_addr(lW2);
    for (int i = tid; i < LW1_H / 8; i += 256) async_ld16(l1 + i * 16, gW1, i * 16);
    for (int i = tid; i < LW2_H / 8; i += 256) async_ld16(l2 + i * 16, gW2, i * 16);
    if (tid < 128) { lB1[tid] = gB1[tid]; lB2[tid] = gB2[tid]; }
    async_fence();
    __syncthreads();
}

// relu(acc + b1) -> per-wave LDS tile (wave-internal DS is in-order)
static __device__ inline void store_relu_T(const v8f acc[8], const float* lB1,
                                           _Float16* myT, int lane) {
    const int hi = lane >> 4, nloc = lane & 15;
#pragma unroll
    for (int ct = 0; ct < 8; ++ct) {
        float bv = lB1[ct * 16 + nloc];
#pragma unroll
        for (int g = 0; g < 8; ++g)
            myT[(g + hi * 8) * TSTRIDE + ct * 16 + nloc] =
                (_Float16)fmaxf(acc[ct][g] + bv, 0.0f);
    }
    lds_fence();
}

// GEMM2 (K=128) from LDS tile; zero-peeled first k-step (inline-0 SRC2).
static __device__ inline void gemm2(const _Float16* myT, const _Float16* lW2,
                                    int lane, v8f out[8]) {
    const int hi = lane >> 4, koff8 = hi ? 8 : 0, nloc = lane & 15;
    const _Float16* trow = myT + nloc * TSTRIDE;
    v8f zero = {0.f, 0.f, 0.f, 0.f, 0.f, 0.f, 0.f, 0.f};
    {
        v16h av = load_a(trow, koff8);
#pragma unroll
        for (int ct = 0; ct < 8; ++ct) {
            v16h bv = *(const v16h*)(lW2 + ((ct * 32 + lane) << 4));
            out[ct] = wmma16(av, bv, zero);
        }
    }
#pragma unroll
    for (int ks = 1; ks < 4; ++ks) {
        v16h av = load_a(trow + ks * 32, koff8);
#pragma unroll
        for (int ct = 0; ct < 8; ++ct) {
            v16h bv = *(const v16h*)(lW2 + (((ks * 8 + ct) * 32 + lane) << 4));
            out[ct] = wmma16(av, bv, out[ct]);
        }
    }
}

// one direction's tail: relu->T, GEMM2, mask+bias accumulate into fin
static __device__ inline void edge_tail(const v8f acc[8], int dir, int rowbase,
                                        const _Float16* lW2, const float* lB1,
                                        const float* lB2, _Float16* myT,
                                        int lane, v8f fin[8]) {
    const int hi = lane >> 4, nloc = lane & 15;
    store_relu_T(acc, lB1, myT, lane);
    v8f out[8];
    gemm2(myT, lW2, lane, out);
#pragma unroll
    for (int ct = 0; ct < 8; ++ct) {
        float bv2 = lB2[ct * 16 + nloc];
#pragma unroll
        for (int g = 0; g < 8; ++g) {
            int trow = rowbase + g + hi * 8;
            float vm = (dir == 0) ? (trow > 0 ? 1.0f : 0.0f)
                                  : (trow < Sn - 1 ? 1.0f : 0.0f);
            float v = vm * (out[ct][g] + bv2);
            if (dir == 0) fin[ct][g] = v;
            else          fin[ct][g] += v;
        }
    }
}

// ---------------------------------------------------------------------------
// Edge kernel: agg[s] = [s>0]*MLP_e([h[s-1],h[s]]) + [s<S-1]*MLP_e([h[s+1],h[s]])
// Both directions share W1 B-fragments: one dual-accumulator GEMM1 pass.
// ---------------------------------------------------------------------------
__global__ void __launch_bounds__(256)
edge_kernel(const _Float16* __restrict__ h16, _Float16* __restrict__ agg,
            const _Float16* __restrict__ gW1, const _Float16* __restrict__ gW2,
            const float* __restrict__ gB1, const float* __restrict__ gB2) {
    extern __shared__ char smem[];
    _Float16* lW1 = (_Float16*)smem;
    _Float16* lW2 = lW1 + LW1_H;
    _Float16* lT  = lW2 + LW2_H;
    float*    lB1 = (float*)(lT + T_H);
    float*    lB2 = lB1 + 128;

    const int tid = threadIdx.x;
    fill_lds(gW1, gW2, gB1, gB2, lW1, lW2, lB1, lB2, tid);

    const int b       = blockIdx.x >> 2;
    const int tile    = blockIdx.x & 3;
    const int wave    = tid >> 5;
    const int lane    = tid & 31;
    const int hi      = lane >> 4;
    const int koff8   = hi ? 8 : 0;
    const int nloc    = lane & 15;
    const int rowbase = tile * 128 + wave * 16;
    const int token_a = rowbase + nloc;
    _Float16* myT = lT + wave * 16 * TSTRIDE;

    const _Float16* rowS  = h16 + ((size_t)b * Sn + token_a) * Hn;
    int nm = token_a > 0 ? token_a - 1 : 0;              // clamped (masked later)
    int np = token_a < Sn - 1 ? token_a + 1 : Sn - 1;
    const _Float16* rowNm = h16 + ((size_t)b * Sn + nm) * Hn;
    const _Float16* rowNp = h16 + ((size_t)b * Sn + np) * Hn;

    v8f acc0[8], acc1[8];
    v8f zero = {0.f, 0.f, 0.f, 0.f, 0.f, 0.f, 0.f, 0.f};
    {   // ks = 0 peeled: inline-0 C operand instead of bias broadcast movs
        v16h a0 = load_a(rowNm, koff8);
        v16h a1 = load_a(rowNp, koff8);
#pragma unroll
        for (int ct = 0; ct < 8; ++ct) {
            v16h bv = *(const v16h*)(lW1 + ((ct * 32 + lane) << 4));
            acc0[ct] = wmma16(a0, bv, zero);
            acc1[ct] = wmma16(a1, bv, zero);
        }
    }
#pragma unroll
    for (int ks = 1; ks < 8; ++ks) {
        v16h a0, a1;
        if (ks < 4) {                  // K 32..127: neighbor rows (per dir)
            a0 = load_a(rowNm + ks * 32, koff8);
            a1 = load_a(rowNp + ks * 32, koff8);
        } else {                       // K 128..255: self row (shared)
            a0 = load_a(rowS + (ks - 4) * 32, koff8);
            a1 = a0;
        }
#pragma unroll
        for (int ct = 0; ct < 8; ++ct) {
            v16h bv = *(const v16h*)(lW1 + (((ks * 8 + ct) * 32 + lane) << 4));
            acc0[ct] = wmma16(a0, bv, acc0[ct]);
            acc1[ct] = wmma16(a1, bv, acc1[ct]);
        }
    }

    v8f fin[8];
    edge_tail(acc0, 0, rowbase, lW2, lB1, lB2, myT, lane, fin);
    edge_tail(acc1, 1, rowbase, lW2, lB1, lB2, myT, lane, fin);

    // fin -> LDS tile (f16), then contiguous vector stores to agg
#pragma unroll
    for (int ct = 0; ct < 8; ++ct)
#pragma unroll
        for (int g = 0; g < 8; ++g)
            myT[(g + hi * 8) * TSTRIDE + ct * 16 + nloc] = (_Float16)fin[ct][g];
    lds_fence();

    const int r = lane >> 1, hp = lane & 1;              // lane pair <-> row
    const _Float16* srcp = myT + r * TSTRIDE + hp * 64;
    _Float16* dstp = agg + ((size_t)b * Sn + rowbase + r) * Hn + hp * 64;
#pragma unroll
    for (int c = 0; c < 4; ++c)
        *(v8h*)(dstp + c * 8) = *(const v8h*)(srcp + c * 8);
}

// ---------------------------------------------------------------------------
// Node kernel: h += MLP_n([h, agg]) * mask ; refresh f16 mirror.
// ---------------------------------------------------------------------------
__global__ void __launch_bounds__(256)
node_kernel(float* __restrict__ h32, _Float16* __restrict__ h16,
            const _Float16* __restrict__ agg,
            const _Float16* __restrict__ gW1, const _Float16* __restrict__ gW2,
            const float* __restrict__ gB1, const float* __restrict__ gB2,
            const float* __restrict__ mask) {
    extern __shared__ char smem[];
    _Float16* lW1 = (_Float16*)smem;
    _Float16* lW2 = lW1 + LW1_H;
    _Float16* lT  = lW2 + LW2_H;
    float*    lB1 = (float*)(lT + T_H);
    float*    lB2 = lB1 + 128;

    const int tid = threadIdx.x;
    fill_lds(gW1, gW2, gB1, gB2, lW1, lW2, lB1, lB2, tid);

    const int b       = blockIdx.x >> 2;
    const int tile    = blockIdx.x & 3;
    const int wave    = tid >> 5;
    const int lane    = tid & 31;
    const int hi      = lane >> 4;
    const int koff8   = hi ? 8 : 0;
    const int nloc    = lane & 15;
    const int rowbase = tile * 128 + wave * 16;
    const int token_a = rowbase + nloc;
    _Float16* myT = lT + wave * 16 * TSTRIDE;

    const _Float16* rowH = h16 + ((size_t)b * Sn + token_a) * Hn;
    const _Float16* rowA = agg + ((size_t)b * Sn + token_a) * Hn;

    v8f acc[8];
    v8f zero = {0.f, 0.f, 0.f, 0.f, 0.f, 0.f, 0.f, 0.f};
    {
        v16h av = load_a(rowH, koff8);
#pragma unroll
        for (int ct = 0; ct < 8; ++ct) {
            v16h bv = *(const v16h*)(lW1 + ((ct * 32 + lane) << 4));
            acc[ct] = wmma16(av, bv, zero);
        }
    }
#pragma unroll
    for (int ks = 1; ks < 8; ++ks) {
        const _Float16* src = (ks < 4 ? rowH : rowA - 128) + (ks & 3) * 32;
        v16h av = load_a(src, koff8);
#pragma unroll
        for (int ct = 0; ct < 8; ++ct) {
            v16h bv = *(const v16h*)(lW1 + (((ks * 8 + ct) * 32 + lane) << 4));
            acc[ct] = wmma16(av, bv, acc[ct]);
        }
    }

    store_relu_T(acc, lB1, myT, lane);
    v8f out[8];
    gemm2(myT, lW2, lane, out);

    // upd (= out + b2) -> LDS tile f16, then contiguous vectorized h update
#pragma unroll
    for (int ct = 0; ct < 8; ++ct) {
        float bv2 = lB2[ct * 16 + nloc];
#pragma unroll
        for (int g = 0; g < 8; ++g)
            myT[(g + hi * 8) * TSTRIDE + ct * 16 + nloc] =
                (_Float16)(out[ct][g] + bv2);
    }
    lds_fence();

    const int r = lane >> 1, hp = lane & 1;
    const _Float16* up = myT + r * TSTRIDE + hp * 64;
    const size_t base = ((size_t)b * Sn + rowbase + r) * Hn + hp * 64;
    const float mv = mask[(size_t)b * Sn + rowbase + r];
    float*    H32 = h32 + base;
    _Float16* H16 = h16 + base;
    _Float16  tmp[64];
#pragma unroll
    for (int c = 0; c < 64; c += 4) {
        v4f hv = *(v4f*)(H32 + c);
#pragma unroll
        for (int j = 0; j < 4; ++j) {
            float nh = hv[j] + (float)up[c + j] * mv;
            hv[j] = nh;
            tmp[c + j] = (_Float16)nh;
        }
        *(v4f*)(H32 + c) = hv;
    }
#pragma unroll
    for (int c = 0; c < 64; c += 8)
        *(v8h*)(H16 + c) = *(const v8h*)(tmp + c);
}

// ---------------------------------------------------------------------------
// Fused masked mean-pool + readout MLP (~17 MFLOP total -> plain VALU).
// ---------------------------------------------------------------------------
__global__ void __launch_bounds__(128)
pool_readout_kernel(const float* __restrict__ h32, const float* __restrict__ mask,
                    const float* __restrict__ w1, const float* __restrict__ b1,
                    const float* __restrict__ w2, const float* __restrict__ b2,
                    float* __restrict__ out) {
    __shared__ float pooled[Hn];
    __shared__ float red[Hn];
    int b = blockIdx.x, k = threadIdx.x;
    float s = 0.0f, msum = 0.0f;
    for (int t = 0; t < Sn; ++t) {
        float mv = mask[(size_t)b * Sn + t];
        s    += h32[((size_t)b * Sn + t) * Hn + k] * mv;
        msum += mv;
    }
    pooled[k] = s / fmaxf(msum, 1.0f);
    __syncthreads();
    float dot = b1[k];
    for (int j = 0; j < Hn; ++j) dot += pooled[j] * w1[(size_t)j * Hn + k];
    red[k] = fmaxf(dot, 0.0f) * w2[k];
    __syncthreads();
    for (int off = 64; off > 0; off >>= 1) {
        if (k < off) red[k] += red[k + off];
        __syncthreads();
    }
    if (k == 0) out[b] = red[0] + b2[0];
}

// ---------------------------------------------------------------------------
// Host side
// ---------------------------------------------------------------------------
extern "C" void kernel_launch(void* const* d_in, const int* in_sizes, int n_in,
                              void* d_out, int out_size, void* d_ws, size_t ws_size,
                              hipStream_t stream) {
    (void)in_sizes; (void)n_in; (void)out_size; (void)ws_size;
    const int*   token_ids = (const int*)  d_in[0];
    const float* mask      = (const float*)d_in[1];
    const float* aa_embed  = (const float*)d_in[2];
    const float* pos_embed = (const float*)d_in[3];
    const float* edge_w1   = (const float*)d_in[4];
    const float* edge_b1   = (const float*)d_in[5];
    const float* edge_w2   = (const float*)d_in[6];
    const float* edge_b2   = (const float*)d_in[7];
    const float* node_w1   = (const float*)d_in[8];
    const float* node_b1   = (const float*)d_in[9];
    const float* node_w2   = (const float*)d_in[10];
    const float* node_b2   = (const float*)d_in[11];
    const float* ro_w1     = (const float*)d_in[12];
    const float* ro_b1     = (const float*)d_in[13];
    const float* ro_w2     = (const float*)d_in[14];
    const float* ro_b2     = (const float*)d_in[15];

    char* ws = (char*)d_ws;
    constexpr size_t OFF_H32 = 0;
    constexpr size_t OFF_H16 = OFF_H32 + (size_t)Bn*Sn*Hn*4;
    constexpr size_t OFF_AGG = OFF_H16 + (size_t)Bn*Sn*Hn*2;
    constexpr size_t OFF_EW1 = OFF_AGG + (size_t)Bn*Sn*Hn*2;
    constexpr size_t OFF_EW2 = OFF_EW1 + (size_t)Ln*256*128*2;
    constexpr size_t OFF_NW1 = OFF_EW2 + (size_t)Ln*128*128*2;
    constexpr size_t OFF_NW2 = OFF_NW1 + (size_t)Ln*256*128*2;

    float*    h32  = (float*)   (ws + OFF_H32);
    _Float16* h16  = (_Float16*)(ws + OFF_H16);
    _Float16* agg  = (_Float16*)(ws + OFF_AGG);
    _Float16* ew1f = (_Float16*)(ws + OFF_EW1);
    _Float16* ew2f = (_Float16*)(ws + OFF_EW2);
    _Float16* nw1f = (_Float16*)(ws + OFF_NW1);
    _Float16* nw2f = (_Float16*)(ws + OFF_NW2);

    wfrag_kernel<<<32, 256, 0, stream>>>(edge_w1, ew1f, 256, Ln);
    wfrag_kernel<<<16, 256, 0, stream>>>(edge_w2, ew2f, 128, Ln);
    wfrag_kernel<<<32, 256, 0, stream>>>(node_w1, nw1f, 256, Ln);
    wfrag_kernel<<<16, 256, 0, stream>>>(node_w2, nw2f, 128, Ln);

    embed_kernel<<<(Bn * Sn * (Hn / 4)) / 256, 256, 0, stream>>>(
        token_ids, mask, aa_embed, pos_embed, h32, h16);

    const int gemm_blocks = Bn * (Sn / 128);   // 2048
    for (int l = 0; l < Ln; ++l) {
        edge_kernel<<<gemm_blocks, 256, SMEM_BYTES, stream>>>(
            h16, agg,
            ew1f + (size_t)l * 256 * 128, ew2f + (size_t)l * 128 * 128,
            edge_b1 + (size_t)l * Hn, edge_b2 + (size_t)l * Hn);
        node_kernel<<<gemm_blocks, 256, SMEM_BYTES, stream>>>(
            h32, h16, agg,
            nw1f + (size_t)l * 256 * 128, nw2f + (size_t)l * 128 * 128,
            node_b1 + (size_t)l * Hn, node_b2 + (size_t)l * Hn, mask);
    }

    pool_readout_kernel<<<Bn, 128, 0, stream>>>(
        h32, mask, ro_w1, ro_b1, ro_w2, ro_b2, (float*)d_out);
}
